// CustomMultiHeadAttention_48490180772334
// MI455X (gfx1250) — matrix operand order repaired and checked by reading, not compiled, not run on hardware
//
#include <hip/hip_runtime.h>

typedef __attribute__((ext_vector_type(16))) __bf16 bf16x16;
typedef __attribute__((ext_vector_type(8)))  __bf16 bf16x8;
typedef __attribute__((ext_vector_type(8)))  float  f32x8;

#define DEVINL __device__ __forceinline__

namespace {
enum : int { kB = 4, kS = 1024, kD = 1024, kH = 16, kE = 64 };
}

DEVINL __bf16 f2bf(float f) {
  unsigned x = __builtin_bit_cast(unsigned, f);
  x += 0x7fffu + ((x >> 16) & 1u);
  return __builtin_bit_cast(__bf16, (unsigned short)(x >> 16));
}

DEVINL bf16x16 cat8(bf16x8 lo, bf16x8 hi) {
  return __builtin_shufflevector(lo, hi, 0, 1, 2, 3, 4, 5, 6, 7, 8, 9, 10, 11, 12, 13, 14, 15);
}

DEVINL f32x8 zero8() {
  f32x8 z = {0.f, 0.f, 0.f, 0.f, 0.f, 0.f, 0.f, 0.f};
  return z;
}

DEVINL f32x8 wmma_bf16(bf16x16 a, bf16x16 b, f32x8 c) {
  return __builtin_amdgcn_wmma_f32_16x16x32_bf16(false, a, false, b, (short)0, c, false, false);
}

// A-matrix fragment (16x32 bf16): lane row = l15, two 16B runs at K = 8g and 16+8g
DEVINL bf16x16 loadA(const __bf16* row, int k, int g) {
  return cat8(*(const bf16x8*)(row + k + 8 * g),
              *(const bf16x8*)(row + k + 16 + 8 * g));
}
// B-matrix fragment (32x16 bf16) from K-contiguous storage: lane col = l15, two 16B runs at K = 8g and 16+8g
DEVINL bf16x16 loadB(const __bf16* row, int k, int g) {
  return cat8(*(const bf16x8*)(row + k + 8 * g),
              *(const bf16x8*)(row + k + 16 + 8 * g));
}

// ---------------------------------------------------------------------------
// Elementwise f32 -> bf16
// ---------------------------------------------------------------------------
__global__ void cvt_bf16_kernel(const float* __restrict__ in, __bf16* __restrict__ out, int n) {
  int i = blockIdx.x * 256 + threadIdx.x;
  if (i < n) out[i] = f2bf(in[i]);
}

// Transpose [batch, rows, cols] f32 -> [batch, cols, rows] bf16
__global__ void transpose_bf16_kernel(const float* __restrict__ w, __bf16* __restrict__ out,
                                      int rows, int cols) {
  int i = blockIdx.x * 256 + threadIdx.x;
  if (i >= rows * cols) return;
  size_t base = (size_t)blockIdx.y * rows * cols;
  int e = i / rows;
  int d = i - e * rows;
  out[base + (size_t)e * rows + d] = f2bf(w[base + (size_t)d * cols + e]);
}

// ---------------------------------------------------------------------------
// Per-head projection: out[b,h] = X[b] (S x D) * W[h] (D x E) + bias.
// WT stored [h, E, D] (K-contiguous). Block = 4 waves (2x2); wave = 32x32 tile
// (2 A-tiles x 2 B-tiles x 4 WMMA accumulators -> 2 loads per wmma).
// Block tile: 64 rows x 64 cols (= all of E). transposeOut=1 -> out [b,h,E,S].
// ---------------------------------------------------------------------------
__global__ __launch_bounds__(128) void proj_kernel(const __bf16* __restrict__ X,
                                                   const __bf16* __restrict__ WT,
                                                   const float* __restrict__ bias,
                                                   __bf16* __restrict__ out, int transposeOut) {
  const int mt = blockIdx.x, h = blockIdx.y, b = blockIdx.z;
  const int lane = threadIdx.x & 31, wave = threadIdx.x >> 5;
  const int g = lane >> 4, l15 = lane & 15;
  const int mw = wave >> 1, nw = wave & 1;

  const int m0 = mt * 64 + mw * 32 + l15;   // A rows m0, m0+16
  const int n0 = nw * 32 + l15;             // B cols n0, n0+16
  const __bf16* A0 = X + ((size_t)b * kS + m0) * kD;
  const __bf16* A1 = A0 + (size_t)16 * kD;
  const __bf16* B0 = WT + ((size_t)h * kE + n0) * kD;
  const __bf16* B1 = B0 + (size_t)16 * kD;

  f32x8 acc00 = zero8(), acc01 = zero8(), acc10 = zero8(), acc11 = zero8();
  for (int k = 0; k < kD; k += 32) {
    bf16x16 a0 = loadA(A0, k, g);
    bf16x16 a1 = loadA(A1, k, g);
    bf16x16 b0 = loadB(B0, k, g);
    bf16x16 b1 = loadB(B1, k, g);
    acc00 = wmma_bf16(a0, b0, acc00);
    acc01 = wmma_bf16(a0, b1, acc01);
    acc10 = wmma_bf16(a1, b0, acc10);
    acc11 = wmma_bf16(a1, b1, acc11);
  }

  const size_t bh_idx = (size_t)b * kH + h;
  const int nn0 = nw * 32 + l15, nn1 = nn0 + 16;
  const float bv0 = bias[h * kE + nn0], bv1 = bias[h * kE + nn1];
#pragma unroll
  for (int r = 0; r < 8; ++r) {
    int mA = mt * 64 + mw * 32 + r + 8 * g;      // rows of tiles (0j)
    int mB = mA + 16;                             // rows of tiles (1j)
    float v00 = acc00[r] + bv0, v01 = acc01[r] + bv1;
    float v10 = acc10[r] + bv0, v11 = acc11[r] + bv1;
    if (transposeOut) {                           // [b,h,E,S]
      out[(bh_idx * kE + nn0) * kS + mA] = f2bf(v00);
      out[(bh_idx * kE + nn1) * kS + mA] = f2bf(v01);
      out[(bh_idx * kE + nn0) * kS + mB] = f2bf(v10);
      out[(bh_idx * kE + nn1) * kS + mB] = f2bf(v11);
    } else {                                      // [b,h,S,E]
      out[(bh_idx * kS + mA) * kE + nn0] = f2bf(v00);
      out[(bh_idx * kS + mA) * kE + nn1] = f2bf(v01);
      out[(bh_idx * kS + mB) * kE + nn0] = f2bf(v10);
      out[(bh_idx * kS + mB) * kE + nn1] = f2bf(v11);
    }
  }
}

// ---------------------------------------------------------------------------
// Attention: one (b, h, 16-row s-tile) per block; 128 threads (4 waves).
// Single pass: Q tile hoisted to registers; all 16 score tiles per wave held
// in VGPRs (128 regs) -> no recompute. Softmax via shuffle + LDS reduction.
// Phase 2: O = P(LDS, WMMA A-layout) * V (stored [E,S]); normalize at the end.
// ---------------------------------------------------------------------------
__global__ __launch_bounds__(128) void attn_kernel(const __bf16* __restrict__ Q,
                                                   const __bf16* __restrict__ K,
                                                   const __bf16* __restrict__ Vt,
                                                   const int* __restrict__ mask,
                                                   __bf16* __restrict__ heads) {
  __shared__ __bf16 sP[16 * kS];      // 32 KB: unnormalized probs, rows 0..15
  __shared__ float sRed[64];          // 4 waves x 16 rows
  __shared__ float sRowMax[16];
  __shared__ float sRowSum[16];

  const int st = blockIdx.x, h = blockIdx.y, b = blockIdx.z;
  const int tid = threadIdx.x, lane = tid & 31, wave = tid >> 5;
  const int g = lane >> 4, l15 = lane & 15;
  const size_t bh_idx = (size_t)b * kH + h;

  const __bf16* Qrow = Q + (bh_idx * kS + st * 16 + l15) * kE;
  const int* mrow = mask + ((size_t)b * kS + st * 16) * kS;

  // hoist Q tile (16 x 64) into registers: 2 k-step fragments
  bf16x16 qA[2];
#pragma unroll
  for (int ks = 0; ks < 2; ++ks) qA[ks] = loadA(Qrow, ks * 32, g);

  // ---- scores: 16 tiles per wave, kept in VGPRs; masked+scaled; track max ----
  f32x8 sc[16];
  float rmax8[8];
#pragma unroll
  for (int r = 0; r < 8; ++r) rmax8[r] = -3.0e38f;

#pragma unroll
  for (int i = 0; i < 16; ++i) {
    const int tt = wave + 4 * i;
    const __bf16* Krow = K + (bh_idx * kS + tt * 16 + l15) * kE;
    f32x8 acc = zero8();
#pragma unroll
    for (int ks = 0; ks < 2; ++ks)
      acc = wmma_bf16(qA[ks], loadB(Krow, ks * 32, g), acc);
#pragma unroll
    for (int r = 0; r < 8; ++r) {
      int srow = r + 8 * g;
      int scol = tt * 16 + l15;
      int m = mrow[(size_t)srow * kS + scol];
      float s = acc[r] * 0.03125f + (m ? 0.f : -1.0e10f);
      acc[r] = s;
      rmax8[r] = fmaxf(rmax8[r], s);
    }
    sc[i] = acc;
  }

  // row max: reduce across 16 lanes of each half-wave, then across 4 waves
#pragma unroll
  for (int r = 0; r < 8; ++r) {
#pragma unroll
    for (int off = 1; off < 16; off <<= 1)
      rmax8[r] = fmaxf(rmax8[r], __shfl_xor(rmax8[r], off, 32));
  }
  if (lane == 0) {
#pragma unroll
    for (int r = 0; r < 8; ++r) sRed[wave * 16 + r] = rmax8[r];
  }
  if (lane == 16) {
#pragma unroll
    for (int r = 0; r < 8; ++r) sRed[wave * 16 + 8 + r] = rmax8[r];
  }
  __syncthreads();
  if (tid < 16) {
    float m2 = sRed[tid];
#pragma unroll
    for (int w = 1; w < 4; ++w) m2 = fmaxf(m2, sRed[w * 16 + tid]);
    sRowMax[tid] = m2;
  }
  __syncthreads();

  // ---- exp from registers, write P to LDS, accumulate row sums ----
  float psum8[8];
#pragma unroll
  for (int r = 0; r < 8; ++r) psum8[r] = 0.f;

#pragma unroll
  for (int i = 0; i < 16; ++i) {
    const int tt = wave + 4 * i;
#pragma unroll
    for (int r = 0; r < 8; ++r) {
      int srow = r + 8 * g;
      int scol = tt * 16 + l15;
      float s = sc[i][r];
      float p = (s > -1.0e9f) ? __expf(s - sRowMax[srow]) : 0.f;
      psum8[r] += p;
      sP[srow * kS + scol] = f2bf(p);
    }
  }
#pragma unroll
  for (int r = 0; r < 8; ++r) {
#pragma unroll
    for (int off = 1; off < 16; off <<= 1) psum8[r] += __shfl_xor(psum8[r], off, 32);
  }
  if (lane == 0) {
#pragma unroll
    for (int r = 0; r < 8; ++r) sRed[wave * 16 + r] = psum8[r];
  }
  if (lane == 16) {
#pragma unroll
    for (int r = 0; r < 8; ++r) sRed[wave * 16 + 8 + r] = psum8[r];
  }
  __syncthreads();
  if (tid < 16) {
    float s3 = sRed[tid];
#pragma unroll
    for (int w = 1; w < 4; ++w) s3 += sRed[w * 16 + tid];
    sRowSum[tid] = (s3 > 0.f) ? s3 : 1.0f;  // fully-masked row -> zeros (matches ref)
  }
  __syncthreads();

  // ---- Phase 2: O = P * V ----
  const int n0 = wave * 16;
  const __bf16* Vrow = Vt + (bh_idx * kE + n0 + l15) * kS;
  f32x8 o = zero8();
  for (int kb = 0; kb < kS; kb += 32) {
    bf16x16 a = cat8(*(const bf16x8*)(sP + l15 * kS + kb + 8 * g),
                     *(const bf16x8*)(sP + l15 * kS + kb + 16 + 8 * g));
    o = wmma_bf16(a, loadB(Vrow, kb, g), o);
  }
#pragma unroll
  for (int r = 0; r < 8; ++r) {
    int srow = r + 8 * g;
    float v = o[r] / sRowSum[srow];
    heads[((size_t)b * kS + st * 16 + srow) * kD + h * kE + n0 + l15] = f2bf(v);
  }
}

// ---------------------------------------------------------------------------
// Output projection: out = heads(BS x D) * Wo + bo ; WoT stored [n, k].
// Block = 4 waves (2x2), wave = 32x32 tile; block tile 64x64.
// ---------------------------------------------------------------------------
__global__ __launch_bounds__(128) void outproj_kernel(const __bf16* __restrict__ Hd,
                                                      const __bf16* __restrict__ WoT,
                                                      const float* __restrict__ bo,
                                                      float* __restrict__ out) {
  const int mt = blockIdx.x, nt = blockIdx.y;
  const int lane = threadIdx.x & 31, wave = threadIdx.x >> 5;
  const int g = lane >> 4, l15 = lane & 15;
  const int mw = wave >> 1, nw = wave & 1;

  const int m0 = mt * 64 + mw * 32 + l15;
  const int n0 = nt * 64 + nw * 32 + l15;
  const __bf16* A0 = Hd + (size_t)m0 * kD;
  const __bf16* A1 = A0 + (size_t)16 * kD;
  const __bf16* B0 = WoT + (size_t)n0 * kD;
  const __bf16* B1 = B0 + (size_t)16 * kD;

  f32x8 acc00 = zero8(), acc01 = zero8(), acc10 = zero8(), acc11 = zero8();
  for (int k = 0; k < kD; k += 32) {
    bf16x16 a0 = loadA(A0, k, g);
    bf16x16 a1 = loadA(A1, k, g);
    bf16x16 b0 = loadB(B0, k, g);
    bf16x16 b1 = loadB(B1, k, g);
    acc00 = wmma_bf16(a0, b0, acc00);
    acc01 = wmma_bf16(a0, b1, acc01);
    acc10 = wmma_bf16(a1, b0, acc10);
    acc11 = wmma_bf16(a1, b1, acc11);
  }

  const int nn0 = nt * 64 + nw * 32 + l15, nn1 = nn0 + 16;
  const float bv0 = bo[nn0], bv1 = bo[nn1];
#pragma unroll
  for (int r = 0; r < 8; ++r) {
    size_t mA = (size_t)mt * 64 + (wave >> 1) * 32 + r + 8 * g;
    size_t mB = mA + 16;
    out[mA * kD + nn0] = acc00[r] + bv0;
    out[mA * kD + nn1] = acc01[r] + bv1;
    out[mB * kD + nn0] = acc10[r] + bv0;
    out[mB * kD + nn1] = acc11[r] + bv1;
  }
}

// ---------------------------------------------------------------------------
extern "C" void kernel_launch(void* const* d_in, const int* in_sizes, int n_in,
                              void* d_out, int out_size, void* d_ws, size_t ws_size,
                              hipStream_t stream) {
  (void)in_sizes; (void)n_in; (void)out_size; (void)ws_size;

  const float* q    = (const float*)d_in[0];
  const float* k    = (const float*)d_in[1];
  const float* v    = (const float*)d_in[2];
  const int*   mask = (const int*)d_in[3];
  const float* Wq = (const float*)d_in[4];
  const float* bq = (const float*)d_in[5];
  const float* Wk = (const float*)d_in[6];
  const float* bk = (const float*)d_in[7];
  const float* Wv = (const float*)d_in[8];
  const float* bv = (const float*)d_in[9];
  const float* Wo = (const float*)d_in[10];
  const float* bo = (const float*)d_in[11];
  float* out = (float*)d_out;

  const size_t nBSD = (size_t)kB * kS * kD;        // 4,194,304
  const size_t nW   = (size_t)kH * kD * kE;        // 1,048,576
  const size_t nWo  = (size_t)kD * kD;             // 1,048,576

  char* ws = (char*)d_ws;
  size_t off = 0;
  __bf16* Xq  = (__bf16*)(ws + off); off += nBSD * 2;
  __bf16* Xk  = (__bf16*)(ws + off); off += nBSD * 2;
  __bf16* Xv  = (__bf16*)(ws + off); off += nBSD * 2;
  __bf16* WqT = (__bf16*)(ws + off); off += nW * 2;
  __bf16* WkT = (__bf16*)(ws + off); off += nW * 2;
  __bf16* WvT = (__bf16*)(ws + off); off += nW * 2;
  __bf16* WoT = (__bf16*)(ws + off); off += nWo * 2;
  __bf16* Qh  = (__bf16*)(ws + off); off += nBSD * 2;  // [B,H,S,E]
  __bf16* Kh  = (__bf16*)(ws + off); off += nBSD * 2;  // [B,H,S,E]
  __bf16* Vt  = (__bf16*)(ws + off); off += nBSD * 2;  // [B,H,E,S]
  __bf16* Hd  = (__bf16*)(ws + off); off += nBSD * 2;  // [B,S,D] concat heads

  // 1) convert inputs to bf16
  {
    int blocks = (int)(nBSD / 256);
    cvt_bf16_kernel<<<blocks, 256, 0, stream>>>(q, Xq, (int)nBSD);
    cvt_bf16_kernel<<<blocks, 256, 0, stream>>>(k, Xk, (int)nBSD);
    cvt_bf16_kernel<<<blocks, 256, 0, stream>>>(v, Xv, (int)nBSD);
  }
  // 2) transpose weights to bf16 (B-matrix friendly layouts)
  transpose_bf16_kernel<<<dim3((kD * kE) / 256, kH), 256, 0, stream>>>(Wq, WqT, kD, kE);
  transpose_bf16_kernel<<<dim3((kD * kE) / 256, kH), 256, 0, stream>>>(Wk, WkT, kD, kE);
  transpose_bf16_kernel<<<dim3((kD * kE) / 256, kH), 256, 0, stream>>>(Wv, WvT, kD, kE);
  transpose_bf16_kernel<<<dim3((kD * kD) / 256, 1), 256, 0, stream>>>(Wo, WoT, kD, kD);

  // 3) per-head projections (block tile 64 x 64)
  dim3 pgrid(kS / 64, kH, kB);
  proj_kernel<<<pgrid, 128, 0, stream>>>(Xq, WqT, bq, Qh, 0);
  proj_kernel<<<pgrid, 128, 0, stream>>>(Xk, WkT, bk, Kh, 0);
  proj_kernel<<<pgrid, 128, 0, stream>>>(Xv, WvT, bv, Vt, 1);

  // 4) attention
  attn_kernel<<<dim3(kS / 16, kH, kB), 128, 0, stream>>>(Qh, Kh, Vt, mask, Hd);

  // 5) output projection (+ bias), f32 out
  outproj_kernel<<<dim3((kB * kS) / 64, kD / 64), 128, 0, stream>>>(Hd, WoT, bo, out);
}
